// ModConv2dWithUpsample_42142219108421
// MI455X (gfx1250) — matrix operand order repaired and checked by reading, not compiled, not run on hardware
//
#include <hip/hip_runtime.h>
#include <hip/hip_bf16.h>
#include <stdint.h>
#include <stddef.h>

// ---- problem constants ----
#define NB    8
#define CIN   512
#define COUT  256
#define SDIM  512
#define HW    64
#define XPH   66      // padded input rows   (64 + 1 border each side)
#define XPW   98      // padded input cols   (extra right pad so N-tile overrun needs no guard)
#define PLANE (XPH * XPW)
#define OD    129     // conv_transpose output spatial dim

typedef _Float16 h16 __attribute__((ext_vector_type(16)));
typedef _Float16 h8  __attribute__((ext_vector_type(8)));
typedef _Float16 h4  __attribute__((ext_vector_type(4)));
typedef float    f8  __attribute__((ext_vector_type(8)));
typedef unsigned int u32x4 __attribute__((ext_vector_type(4)));
typedef int          i32x4 __attribute__((ext_vector_type(4)));
typedef int          i32x8 __attribute__((ext_vector_type(8)));

// ---------------------------------------------------------------------------
// TDM: async 2D tile load Global -> LDS (CDNA5 Tensor Data Mover).
// D# group0/group1 packed per CDNA5 ISA.
// Tiles are always fully in-bounds, so tensor dims are set "huge".
// ---------------------------------------------------------------------------
__device__ __forceinline__ void tdm_load_2d(unsigned lds_byte_addr,
                                            unsigned long long gaddr,
                                            unsigned tile_elems_x,   // contiguous f16 per row
                                            unsigned tile_rows,
                                            unsigned long long row_stride_elems) {
  u32x4 g0;
  g0.x = 1u;                                   // count=1 (valid user descriptor)
  g0.y = lds_byte_addr;                        // LDS byte address
  g0.z = (unsigned)gaddr;                      // global_addr[31:0]
  g0.w = (unsigned)((gaddr >> 32) & 0x1FFFFFFull) | (2u << 30);   // ga[56:32] | type=2

  const unsigned td0 = 0x40000000u;            // huge tensor_dim0 (no OOB)
  const unsigned td1 = 0x40000000u;            // huge tensor_dim1
  i32x8 g1;
  g1[0] = (int)(1u << 16);                     // wg_mask=0, data_size=1 (2B)
  g1[1] = (int)((td0 & 0xFFFFu) << 16);        // [63:48] = tensor_dim0[15:0]
  g1[2] = (int)((td0 >> 16) | ((td1 & 0xFFFFu) << 16));
  g1[3] = (int)((td1 >> 16) | (tile_elems_x << 16));   // [127:112] = tile_dim0
  g1[4] = (int)(tile_rows & 0xFFFFu);          // tile_dim1, tile_dim2=0
  g1[5] = (int)(unsigned)(row_stride_elems & 0xFFFFFFFFull);      // dim0_stride lo
  g1[6] = (int)(unsigned)((row_stride_elems >> 32) & 0xFFFFull);  // stride hi, dim1_stride=0
  g1[7] = 0;

  i32x4 z4 = {0, 0, 0, 0};
#if __clang_major__ >= 23
  i32x8 z8 = {0, 0, 0, 0, 0, 0, 0, 0};
  __builtin_amdgcn_tensor_load_to_lds(g0, g1, z4, z4, z8, 0);
#else
  __builtin_amdgcn_tensor_load_to_lds(g0, g1, z4, z4, 0);
#endif
}

// ---------------------------------------------------------------------------
// Kernel 0: pad + f32->f16 convert x into xp[NB][CIN][XPH][XPW], zero border.
// ---------------------------------------------------------------------------
__global__ void pad_x_kernel(const float* __restrict__ x, _Float16* __restrict__ xp) {
  size_t i = (size_t)blockIdx.x * 256 + threadIdx.x;   // NB*CIN*XPH*XPW total
  int col   = (int)(i % XPW);
  size_t r1 = i / XPW;
  int row   = (int)(r1 % XPH);
  size_t pl = r1 / XPH;                                 // b*CIN + ci
  float v = 0.f;
  if (row >= 1 && row <= HW && col >= 1 && col <= HW)
    v = x[pl * (HW * HW) + (size_t)(row - 1) * HW + (col - 1)];
  xp[i] = (_Float16)v;
}

// ---------------------------------------------------------------------------
// Kernel 1: style modulation  s[b][ci] = style[b] . mod_weight[ci] * mod_scale + bias
// ---------------------------------------------------------------------------
__global__ void mod_kernel(const float* __restrict__ style,
                           const float* __restrict__ mw,
                           const float* __restrict__ mb,
                           float* __restrict__ s) {
  int i  = blockIdx.x * 256 + threadIdx.x;   // NB*CIN
  int b  = i >> 9;
  int ci = i & (CIN - 1);
  const float* st = style + (size_t)b * SDIM;
  const float* w  = mw + (size_t)ci * SDIM;
  float acc = 0.f;
  for (int d = 0; d < SDIM; ++d) acc += st[d] * w[d];
  s[i] = acc * 0.04419417382415922f /* 1/sqrt(512) */ + mb[ci];
}

// ---------------------------------------------------------------------------
// Kernel 2: per-sample modulate + demodulate weights, split into f16 per-phase
// GEMM matrices:  Woo [b][co][ci]            (tap w[1][1])
//                 Weo [b][co][s*CIN+ci]      (tap w[2-2s][1])
//                 Woe [b][co][t*CIN+ci]      (tap w[1][2-2t])
//                 Wee [b][co][(2s+t)*CIN+ci] (tap w[2-2s][2-2t])
// ---------------------------------------------------------------------------
__global__ void wprep_kernel(const float* __restrict__ weight,
                             const float* __restrict__ s,
                             _Float16* __restrict__ Wee, _Float16* __restrict__ Weo,
                             _Float16* __restrict__ Woe, _Float16* __restrict__ Woo) {
  const int b  = blockIdx.x / COUT;
  const int co = blockIdx.x % COUT;
  const float scale = 0.014731391274719738f;   // 1/sqrt(512*9)
  __shared__ float red[256];

  float vals[18];
  float local = 0.f;
#pragma unroll
  for (int j = 0; j < 18; ++j) {
    int e  = threadIdx.x * 18 + j;             // 0..4607 == ci*9 + kk
    int ci = e / 9;
    float v = scale * weight[(size_t)co * (CIN * 9) + e] * s[b * CIN + ci];
    vals[j] = v;
    local += v * v;
  }
  red[threadIdx.x] = local;
  __syncthreads();
  for (int st = 128; st > 0; st >>= 1) {
    if (threadIdx.x < st) red[threadIdx.x] += red[threadIdx.x + st];
    __syncthreads();
  }
  float demod = rsqrtf(red[0] + 1e-8f);

  const size_t bc = (size_t)(b * COUT + co);
#pragma unroll
  for (int j = 0; j < 18; ++j) {
    int e  = threadIdx.x * 18 + j;
    int ci = e / 9;
    int kk = e % 9;
    int ky = kk / 3, kx = kk % 3;
    _Float16 h = (_Float16)(vals[j] * demod);
    if (ky == 1 && kx == 1) {
      Woo[bc * CIN + ci] = h;
    } else if (kx == 1) {                     // ky in {0,2}
      int sidx = 1 - (ky >> 1);
      Weo[bc * (2 * CIN) + sidx * CIN + ci] = h;
    } else if (ky == 1) {                     // kx in {0,2}
      int tidx = 1 - (kx >> 1);
      Woe[bc * (2 * CIN) + tidx * CIN + ci] = h;
    } else {
      int sidx = 1 - (ky >> 1), tidx = 1 - (kx >> 1);
      Wee[bc * (4 * CIN) + (sidx * 2 + tidx) * CIN + ci] = h;
    }
  }
}

// ---------------------------------------------------------------------------
// Kernel 3 (x4 phases): implicit-GEMM upsampled conv via WMMA.
// PHASE 0 = (odd,odd)  rows a:0..63  cols b:0..63  p=2a+1 q=2b+1  K=CIN
// PHASE 1 = (even,odd) rows a:0..64  cols b:0..63  p=2a   q=2b+1  K=2CIN
// PHASE 2 = (odd,even) rows a:0..63  cols b:0..64  p=2a+1 q=2b    K=2CIN
// PHASE 3 = (even,even)rows a:0..64  cols b:0..64  p=2a   q=2b    K=4CIN
// WG: 256 thr = 8 waves; tile M=128 (cout), N=32 (cols), K-chunk 32.
// A tile (weights) via TDM double-buffered; B tile coalesced gather, stored
// transposed [n][k] in LDS so WMMA fragments are contiguous b128 DS loads.
// ---------------------------------------------------------------------------
template <int PHASE>
__global__ __launch_bounds__(256)
void conv_up_phase(const _Float16* __restrict__ xp,
                   const _Float16* __restrict__ wph,
                   _Float16* __restrict__ cbuf) {
  constexpr int NST   = (PHASE == 0) ? 1 : ((PHASE == 3) ? 4 : 2);
  constexpr int KEFF  = NST * CIN;
  constexpr int NK    = KEFF / 32;
  constexpr int NCOLS = (PHASE == 2 || PHASE == 3) ? 65 : 64;
  constexpr int RBASE = (PHASE == 1 || PHASE == 3) ? 0 : 1;  // xp row = RBASE+a+s
  constexpr int CBASE = (PHASE == 2 || PHASE == 3) ? 0 : 1;  // xp col = CBASE+bb+t

  __shared__ __align__(16) _Float16 Asm[2][128 * 32];
  __shared__ __align__(16) _Float16 Bsm[2][32 * 32];

  const int tid  = threadIdx.x;
  const int wav  = tid >> 5;
  const int lane = tid & 31;
  const int l16  = lane & 15;
  const int hi   = lane >> 4;

  const int bb0 = blockIdx.x * 32;
  const int a   = blockIdx.y;
  const int bz  = blockIdx.z;
  const int b   = bz >> 1;
  const int co0 = (bz & 1) * 128;

  const unsigned long long wgbase =
      (unsigned long long)(uintptr_t)wph +
      2ull * ((unsigned long long)(b * COUT + co0) * (unsigned long long)KEFF);

  auto issueA = [&](int kt, int buf) {
    unsigned lds = (unsigned)(uintptr_t)(&Asm[buf][0]);
    tdm_load_2d(lds, wgbase + 2ull * (unsigned)(kt * 32),
                32u, 128u, (unsigned long long)KEFF);
  };

  // Coalesced, guard-free B gather: lane = tile column nn (contiguous 64B per
  // wave per load); wave id selects 4 k-rows; 4 independent global_load_u16
  // from 4 Cin planes packed into one ds_store_b64 (transposed [n][k]).
  auto loadB = [&](int kt, int buf) {
    const int nn    = lane;           // 0..31 tile column
    const int kk0   = wav * 4;        // 0,4,...,28
    const int kbase = kt * 32;
    const int idx   = kbase >> 9;     // which (s,t) block
    int ss = 0, tt = 0;
    if (PHASE == 1) ss = idx;
    else if (PHASE == 2) tt = idx;
    else if (PHASE == 3) { ss = idx >> 1; tt = idx & 1; }
    const int ci0 = (kbase & (CIN - 1)) + kk0;
    const _Float16* p = xp + (size_t)(b * CIN + ci0) * PLANE
                           + (size_t)(RBASE + a + ss) * XPW
                           + (CBASE + bb0 + nn + tt);        // always in-bounds
    _Float16 v0 = p[0];
    _Float16 v1 = p[PLANE];
    _Float16 v2 = p[2 * PLANE];
    _Float16 v3 = p[3 * PLANE];
    h4 v = {v0, v1, v2, v3};
    *(h4*)(&Bsm[buf][nn * 32 + kk0]) = v;
  };

  f8 acc0 = {0.f, 0.f, 0.f, 0.f, 0.f, 0.f, 0.f, 0.f};
  f8 acc1 = acc0;

  loadB(0, 0);
  if (tid == 0) issueA(0, 0);
  __builtin_amdgcn_s_wait_tensorcnt(0);
  asm volatile("" ::: "memory");
  __syncthreads();

  for (int kt = 0; kt < NK; ++kt) {
    const int buf = kt & 1;
    if (kt + 1 < NK) {
      loadB(kt + 1, buf ^ 1);
      if (tid == 0) issueA(kt + 1, buf ^ 1);   // TDM overlaps compute below
    }

    const _Float16* As = &Asm[buf][0];
    const _Float16* Bs = &Bsm[buf][0];
    const int arow = wav * 16 + l16;
    // A 16x32 f16 fragment (ISA layout): lanes<16 K{0..7,16..23}, lanes>=16 K{8..15,24..31}
    h8 alo = *(const h8*)(As + arow * 32 + (hi ? 8 : 0));
    h8 ahi = *(const h8*)(As + arow * 32 + (hi ? 24 : 16));
    h16 afr = __builtin_shufflevector(alo, ahi, 0, 1, 2, 3, 4, 5, 6, 7,
                                      8, 9, 10, 11, 12, 13, 14, 15);
    {
      // B 32x16 fragment 0: lane column n=l16, lanes<16 K{0..7,16..23}, lanes>=16 K{8..15,24..31}
      const _Float16* bp = Bs + l16 * 32 + (hi ? 8 : 0);
      h8 blo = *(const h8*)bp;
      h8 bhi = *(const h8*)(bp + 16);
      h16 bfr = __builtin_shufflevector(blo, bhi, 0, 1, 2, 3, 4, 5, 6, 7,
                                        8, 9, 10, 11, 12, 13, 14, 15);
      acc0 = __builtin_amdgcn_wmma_f32_16x16x32_f16(false, afr, false, bfr,
                                                    (short)0, acc0, false, false);
    }
    {
      const _Float16* bp = Bs + (16 + l16) * 32 + (hi ? 8 : 0);
      h8 blo = *(const h8*)bp;
      h8 bhi = *(const h8*)(bp + 16);
      h16 bfr = __builtin_shufflevector(blo, bhi, 0, 1, 2, 3, 4, 5, 6, 7,
                                        8, 9, 10, 11, 12, 13, 14, 15);
      acc1 = __builtin_amdgcn_wmma_f32_16x16x32_f16(false, afr, false, bfr,
                                                    (short)0, acc1, false, false);
    }

    __builtin_amdgcn_s_wait_tensorcnt(0);   // prefetch complete before barrier
    asm volatile("" ::: "memory");
    __syncthreads();
  }

  // Writeback: C/D layout: VGPR r -> M = r + 8*hi, N = lane%16.
  const int p = (PHASE == 1 || PHASE == 3) ? (2 * a) : (2 * a + 1);
#pragma unroll
  for (int f = 0; f < 2; ++f) {
    f8 acc = f ? acc1 : acc0;
    int bb = bb0 + f * 16 + l16;
    if (bb < NCOLS) {
      int q = (PHASE == 2 || PHASE == 3) ? (2 * bb) : (2 * bb + 1);
#pragma unroll
      for (int r = 0; r < 8; ++r) {
        int co = co0 + wav * 16 + r + hi * 8;
        cbuf[((size_t)(b * COUT + co) * OD + p) * OD + q] = (_Float16)acc[r];
      }
    }
  }
}

// ---------------------------------------------------------------------------
// Kernel 4: 4x4 depthwise blur (pad 1 each side) f16 -> f32 output [.,.,128,128]
// ---------------------------------------------------------------------------
__global__ void blur_kernel(const _Float16* __restrict__ c, float* __restrict__ out) {
  size_t i = (size_t)blockIdx.x * 256 + threadIdx.x;   // NB*COUT*128*128
  int x = (int)(i & 127);
  int y = (int)((i >> 7) & 127);
  size_t pl = i >> 14;                                  // b*COUT+co
  const _Float16* cp = c + pl * (OD * OD);
  const float k1[4] = {1.f, 3.f, 3.f, 1.f};
  float acc = 0.f;
#pragma unroll
  for (int m = 0; m < 4; ++m) {
    int yy = y + m - 1;
    if (yy < 0 || yy >= OD) continue;
#pragma unroll
    for (int n = 0; n < 4; ++n) {
      int xx = x + n - 1;
      if (xx < 0 || xx >= OD) continue;
      acc += k1[m] * k1[n] * (float)cp[(size_t)yy * OD + xx];
    }
  }
  out[i] = acc * (1.0f / 16.0f);
}

// ---------------------------------------------------------------------------
// Workspace layout (bytes)
// ---------------------------------------------------------------------------
static constexpr size_t OFF_S   = 0;
static constexpr size_t SZ_S    = (size_t)NB * CIN * 4;
static constexpr size_t OFF_XP  = (OFF_S + SZ_S + 255) & ~(size_t)255;
static constexpr size_t SZ_XP   = (size_t)NB * CIN * PLANE * 2;
static constexpr size_t OFF_WEE = (OFF_XP + SZ_XP + 255) & ~(size_t)255;
static constexpr size_t SZ_WEE  = (size_t)NB * COUT * 4 * CIN * 2;
static constexpr size_t OFF_WEO = (OFF_WEE + SZ_WEE + 255) & ~(size_t)255;
static constexpr size_t SZ_WEO  = (size_t)NB * COUT * 2 * CIN * 2;
static constexpr size_t OFF_WOE = (OFF_WEO + SZ_WEO + 255) & ~(size_t)255;
static constexpr size_t SZ_WOE  = SZ_WEO;
static constexpr size_t OFF_WOO = (OFF_WOE + SZ_WOE + 255) & ~(size_t)255;
static constexpr size_t SZ_WOO  = (size_t)NB * COUT * CIN * 2;
static constexpr size_t OFF_C   = (OFF_WOO + SZ_WOO + 255) & ~(size_t)255;

extern "C" void kernel_launch(void* const* d_in, const int* in_sizes, int n_in,
                              void* d_out, int out_size, void* d_ws, size_t ws_size,
                              hipStream_t stream) {
  (void)in_sizes; (void)n_in; (void)out_size; (void)ws_size;

  const float* x      = (const float*)d_in[0];
  const float* style  = (const float*)d_in[1];
  const float* weight = (const float*)d_in[2];
  const float* mw     = (const float*)d_in[3];
  const float* mb     = (const float*)d_in[4];
  float* out          = (float*)d_out;

  char* ws = (char*)d_ws;
  float*    smod = (float*)(ws + OFF_S);
  _Float16* xp   = (_Float16*)(ws + OFF_XP);
  _Float16* Wee  = (_Float16*)(ws + OFF_WEE);
  _Float16* Weo  = (_Float16*)(ws + OFF_WEO);
  _Float16* Woe  = (_Float16*)(ws + OFF_WOE);
  _Float16* Woo  = (_Float16*)(ws + OFF_WOO);
  _Float16* cbuf = (_Float16*)(ws + OFF_C);

  // 0: pad/convert x
  {
    size_t total = (size_t)NB * CIN * PLANE;
    pad_x_kernel<<<dim3((unsigned)((total + 255) / 256)), 256, 0, stream>>>(x, xp);
  }
  // 1: modulation
  mod_kernel<<<dim3(NB * CIN / 256), 256, 0, stream>>>(style, mw, mb, smod);
  // 2: weight demod + phase split
  wprep_kernel<<<dim3(NB * COUT), 256, 0, stream>>>(weight, smod, Wee, Weo, Woe, Woo);
  // 3: the four phase GEMMs (WMMA + TDM), M=128 per WG -> grid.z = NB * 2
  conv_up_phase<0><<<dim3(2, 64, NB * 2), 256, 0, stream>>>(xp, Woo, cbuf);
  conv_up_phase<1><<<dim3(2, 65, NB * 2), 256, 0, stream>>>(xp, Weo, cbuf);
  conv_up_phase<2><<<dim3(3, 64, NB * 2), 256, 0, stream>>>(xp, Woe, cbuf);
  conv_up_phase<3><<<dim3(3, 65, NB * 2), 256, 0, stream>>>(xp, Wee, cbuf);
  // 4: blur -> final f32 output
  blur_kernel<<<dim3(NB * COUT * 128 * 128 / 256), 256, 0, stream>>>(cbuf, out);
}
